// RWKV7Block_34815004902093
// MI455X (gfx1250) — matrix operand order repaired and checked by reading, not compiled, not run on hardware
//
#include <hip/hip_runtime.h>
#include <hip/hip_bf16.h>

// Problem constants (match reference)
#define B_  4
#define T_  2048
#define D_  1024
#define H_  16
#define M_  (B_*T_)          // 8192 rows for all GEMMs
static constexpr float LOG_HALF = -0.6065306597126334f;
static constexpr float EPS_GN   = 64.0f * 1e-5f;   // DK * norm_eps

typedef __attribute__((ext_vector_type(16))) __bf16 v16bf;
typedef __attribute__((ext_vector_type(8)))  float  v8f;

union Frag16 { float4 f4[2]; v16bf v; };

__device__ __forceinline__ float sigmoidf_(float x) { return 1.0f / (1.0f + __expf(-x)); }

// ---------------------------------------------------------------------------
// Tensor Data Mover tile loader (gfx1250). Builds a D# per ISA 08_async_tensor
// Group0: count=1 | lds_addr | global_addr[56:0] | type=2
// Group1: data_size=2B, tensor_dim0/1, tile_dim0/1, tensor_dim0_stride
// ---------------------------------------------------------------------------
#ifndef USE_TDM
#if defined(__has_builtin)
#if __has_builtin(__builtin_amdgcn_tensor_load_to_lds) && __has_builtin(__builtin_amdgcn_s_wait_tensorcnt)
#define USE_TDM 1
#else
#define USE_TDM 0
#endif
#else
#define USE_TDM 0
#endif
#endif

#if USE_TDM
typedef unsigned int v4u_t __attribute__((ext_vector_type(4)));
typedef int v8i_t __attribute__((ext_vector_type(8)));
typedef int v4i_t __attribute__((ext_vector_type(4)));

__device__ __forceinline__ void tdm_load_tile_2d(const void* gsrc, unsigned ldsOff,
                                                 unsigned tensorW, unsigned tensorH,
                                                 unsigned tileW, unsigned tileH,
                                                 unsigned strideElems) {
  unsigned long long ga = (unsigned long long)(uintptr_t)gsrc;
  v4u_t g0;
  g0.x = 1u;                                              // count=1 (valid user D#)
  g0.y = ldsOff;                                          // lds_addr (bytes)
  g0.z = (unsigned)ga;                                    // global_addr[31:0]
  g0.w = (unsigned)((ga >> 32) & 0x01FFFFFFu) | (2u << 30); // addr[56:32] | type=2
  v8i_t g1;
  g1[0] = (int)(1u << 16);                                // data_size=1 -> 2 bytes/elem
  g1[1] = (int)((tensorW & 0xFFFFu) << 16);               // tensor_dim0[15:0] @ bits63:48
  g1[2] = (int)((tensorW >> 16) | ((tensorH & 0xFFFFu) << 16)); // dim0 hi | dim1 lo
  g1[3] = (int)((tensorH >> 16) | (tileW << 16));         // dim1 hi | tile_dim0
  g1[4] = (int)(tileH & 0xFFFFu);                         // tile_dim1 (tile_dim2=0)
  g1[5] = (int)strideElems;                               // tensor_dim0_stride[31:0]
  g1[6] = 0;                                              // stride hi | dim1_stride lo
  g1[7] = 0;
  v4i_t z4 = {0, 0, 0, 0};
#if __has_include(<hip/amd_detail/amd_gfx1250_TDM.h>)
  v8i_t z8 = {0, 0, 0, 0, 0, 0, 0, 0};
  __builtin_amdgcn_tensor_load_to_lds(g0, g1, z4, z4, z8, 0);   // clang-23 6-arg form
#else
  __builtin_amdgcn_tensor_load_to_lds(g0, g1, z4, z4, 0);       // ROCm 7.2 5-arg form
#endif
}
#endif

// ---------------------------------------------------------------------------
// Weight transpose + f32->bf16 convert:  Wt[c][r] = bf16(W[r][c])
// ---------------------------------------------------------------------------
__global__ void transcvt_kernel(const float* __restrict__ W,
                                __hip_bfloat16* __restrict__ Wt,
                                int Rows, int Cols) {
  int i = blockIdx.x * blockDim.x + threadIdx.x;
  if (i < Rows * Cols) {
    int r = i / Cols, c = i - r * Cols;
    Wt[(size_t)c * Rows + r] = __float2bfloat16(W[i]);
  }
}

// ---------------------------------------------------------------------------
// Time-shift mixing: x_i = h + (h[t-1]-h)*mix_i ; also query -> bf16
// ---------------------------------------------------------------------------
__global__ void mix_kernel(const float* __restrict__ hs, const float* __restrict__ q,
                           const float* __restrict__ mw, const float* __restrict__ mk,
                           const float* __restrict__ mv, const float* __restrict__ ma,
                           const float* __restrict__ mg,
                           __hip_bfloat16* __restrict__ qb,
                           __hip_bfloat16* __restrict__ xwb, __hip_bfloat16* __restrict__ xkb,
                           __hip_bfloat16* __restrict__ xvb, __hip_bfloat16* __restrict__ xab,
                           __hip_bfloat16* __restrict__ xgb) {
  size_t idx = (size_t)blockIdx.x * blockDim.x + threadIdx.x;
  if (idx >= (size_t)M_ * D_) return;
  int d  = (int)(idx % D_);
  int t  = (int)((idx / D_) % T_);
  float h  = hs[idx];
  float sh = (t > 0) ? hs[idx - D_] : 0.0f;
  float delta = sh - h;
  qb[idx]  = __float2bfloat16(q[idx]);
  xwb[idx] = __float2bfloat16(h + delta * mw[d]);
  xkb[idx] = __float2bfloat16(h + delta * mk[d]);
  xvb[idx] = __float2bfloat16(h + delta * mv[d]);
  xab[idx] = __float2bfloat16(h + delta * ma[d]);
  xgb[idx] = __float2bfloat16(h + delta * mg[d]);
}

// ---------------------------------------------------------------------------
// Elementwise activation + convert to bf16; mode: 0=id, 1=tanh, 2=sigmoid
// ---------------------------------------------------------------------------
__global__ void act_cvt_kernel(const float* __restrict__ x,
                               __hip_bfloat16* __restrict__ y, size_t n, int mode) {
  size_t i = (size_t)blockIdx.x * blockDim.x + threadIdx.x;
  if (i >= n) return;
  float v = x[i];
  if (mode == 1) v = tanhf(v);
  else if (mode == 2) v = sigmoidf_(v);
  y[i] = __float2bfloat16(v);
}

// ---------------------------------------------------------------------------
// WMMA bf16 GEMM:  C[M][N] = A[M][K] * B[K][N]  (B passed transposed Bt[N][K])
// Block tile 128x128, K-step 32, double-buffered LDS. Tiles staged by TDM
// (tensor_load_to_lds, TENSORcnt) with vector-load fallback.
// Requires M % 128 == 0, K % 32 == 0. N guarded.
// ---------------------------------------------------------------------------
__global__ __launch_bounds__(256)
void gemm_bf16_nt(const __hip_bfloat16* __restrict__ A,
                  const __hip_bfloat16* __restrict__ Bt,
                  float* __restrict__ C, int M, int N, int K) {
  __shared__ __align__(16) __hip_bfloat16 sA[2][128 * 32];
  __shared__ __align__(16) __hip_bfloat16 sB[2][128 * 32];

  const int tid  = threadIdx.x;
  const int lane = tid & 31;
  const int wid  = tid >> 5;
  const int wm   = wid & 3;      // wave along M (4 x 32 rows)
  const int wn   = wid >> 2;     // wave along N (2 x 64 cols)
  const int lrow = lane & 15;
  const int lhi  = lane >> 4;
  const int m0   = blockIdx.x * 128;
  const int n0   = blockIdx.y * 128;
  const int nTiles = K >> 5;

  v8f acc[2][4];
#pragma unroll
  for (int i = 0; i < 2; ++i)
#pragma unroll
    for (int j = 0; j < 4; ++j)
#pragma unroll
      for (int e = 0; e < 8; ++e) acc[i][j][e] = 0.0f;

#if USE_TDM
  // ---- prologue: TDM tile 0 into buffer 0 ----
  if (tid == 0) {
    tdm_load_tile_2d(A + (size_t)m0 * K, (unsigned)(uintptr_t)&sA[0][0],
                     32u, 128u, 32u, 128u, (unsigned)K);
    tdm_load_tile_2d(Bt + (size_t)n0 * K, (unsigned)(uintptr_t)&sB[0][0],
                     32u, (unsigned)(N - n0), 32u, 128u, (unsigned)K);
  }
  if (wid == 0) __builtin_amdgcn_s_wait_tensorcnt(0);
  __syncthreads();
#else
  const int ldRow = tid >> 1;
  const int ldCol = (tid & 1) * 16;
  {
    const float4* s4 = reinterpret_cast<const float4*>(A + (size_t)(m0 + ldRow) * K + ldCol);
    float4 a0 = s4[0], a1 = s4[1];
    float4 b0 = make_float4(0.f, 0.f, 0.f, 0.f), b1 = b0;
    if (n0 + ldRow < N) {
      const float4* t4 = reinterpret_cast<const float4*>(Bt + (size_t)(n0 + ldRow) * K + ldCol);
      b0 = t4[0]; b1 = t4[1];
    }
    float4* da = reinterpret_cast<float4*>(&sA[0][0] + ldRow * 32 + ldCol);
    da[0] = a0; da[1] = a1;
    float4* db = reinterpret_cast<float4*>(&sB[0][0] + ldRow * 32 + ldCol);
    db[0] = b0; db[1] = b1;
  }
  __syncthreads();
#endif

  for (int it = 0; it < nTiles; ++it) {
    const int cur = it & 1;
    const int nxt = cur ^ 1;
    const int ktn = (it + 1) << 5;
    const bool haveNext = (it + 1) < nTiles;

#if USE_TDM
    if (haveNext && tid == 0) {
      tdm_load_tile_2d(A + (size_t)m0 * K + ktn, (unsigned)(uintptr_t)&sA[nxt][0],
                       32u, 128u, 32u, 128u, (unsigned)K);
      tdm_load_tile_2d(Bt + (size_t)n0 * K + ktn, (unsigned)(uintptr_t)&sB[nxt][0],
                       32u, (unsigned)(N - n0), 32u, 128u, (unsigned)K);
    }
#else
    float4 a0, a1, b0, b1;
    if (haveNext) {
      const float4* s4 = reinterpret_cast<const float4*>(A + (size_t)(m0 + ldRow) * K + ktn + ldCol);
      a0 = s4[0]; a1 = s4[1];
      b0 = make_float4(0.f, 0.f, 0.f, 0.f); b1 = b0;
      if (n0 + ldRow < N) {
        const float4* t4 = reinterpret_cast<const float4*>(Bt + (size_t)(n0 + ldRow) * K + ktn + ldCol);
        b0 = t4[0]; b1 = t4[1];
      }
    }
#endif

    // ---- LDS -> fragments (ISA 7.12.2 16-bit layouts) + WMMA on current buf ----
    Frag16 af[2], bf[4];
#pragma unroll
    for (int sm = 0; sm < 2; ++sm) {
      int row = wm * 32 + sm * 16 + lrow;
      const float4* p = reinterpret_cast<const float4*>(&sA[cur][0] + row * 32 + lhi * 8);
      af[sm].f4[0] = p[0];   // K = lhi*8 .. +7
      af[sm].f4[1] = p[2];   // K = lhi*8+16 .. +23
    }
#pragma unroll
    for (int sn = 0; sn < 4; ++sn) {
      int row = wn * 64 + sn * 16 + lrow;
      const float4* p = reinterpret_cast<const float4*>(&sB[cur][0] + row * 32 + lhi * 8);
      bf[sn].f4[0] = p[0];   // K = lhi*8 .. +7
      bf[sn].f4[1] = p[2];   // K = lhi*8+16 .. +23
    }
#pragma unroll
    for (int sm = 0; sm < 2; ++sm)
#pragma unroll
      for (int sn = 0; sn < 4; ++sn)
        acc[sm][sn] = __builtin_amdgcn_wmma_f32_16x16x32_bf16(
            false, af[sm].v, false, bf[sn].v, (short)0, acc[sm][sn], false, false);

#if USE_TDM
    if (wid == 0) __builtin_amdgcn_s_wait_tensorcnt(0);
#else
    if (haveNext) {
      float4* da = reinterpret_cast<float4*>(&sA[nxt][0] + ldRow * 32 + ldCol);
      da[0] = a0; da[1] = a1;
      float4* db = reinterpret_cast<float4*>(&sB[nxt][0] + ldRow * 32 + ldCol);
      db[0] = b0; db[1] = b1;
    }
#endif
    __syncthreads();
  }

  // ---- store C (C/D layout: N = lane&15, M = vgpr + 8*(lane>=16)) ----
#pragma unroll
  for (int sm = 0; sm < 2; ++sm) {
    int mBase = m0 + wm * 32 + sm * 16 + lhi * 8;
#pragma unroll
    for (int sn = 0; sn < 4; ++sn) {
      int n = n0 + wn * 64 + sn * 16 + lrow;
      if (n < N) {
#pragma unroll
        for (int r = 0; r < 8; ++r)
          C[(size_t)(mBase + r) * N + n] = acc[sm][sn][r];
      }
    }
  }
}

// ---------------------------------------------------------------------------
// Prep: per (b,t,h) row (one wave32 per row, 2 channels per lane)
// ---------------------------------------------------------------------------
__global__ __launch_bounds__(256)
void prep_kernel(float* __restrict__ kio,   // in: k0, out: kfin
                 float* __restrict__ aio,   // in: a2, out: aterm (-kk)
                 float* __restrict__ wio,   // in: w2, out: ew = exp(w)
                 float* __restrict__ bterm,
                 const float* __restrict__ kkw, const float* __restrict__ kaw,
                 const float* __restrict__ a_bias, const float* __restrict__ w_bias) {
  int lane = threadIdx.x & 31;
  int wid  = threadIdx.x >> 5;
  long row = (long)blockIdx.x * 8 + wid;       // over B*T*H
  int  h   = (int)(row % H_);
  size_t e0 = (size_t)row * 64;
  int d0 = h * 64 + lane, d1 = d0 + 32;
  size_t i0 = e0 + lane,  i1 = i0 + 32;

  float k0a = kio[i0], k0b = kio[i1];
  float kha = k0a * kkw[d0], khb = k0b * kkw[d1];
  float ss = kha * kha + khb * khb;
#pragma unroll
  for (int m = 16; m >= 1; m >>= 1) ss += __shfl_xor(ss, m, 32);
  float inv = 1.0f / fmaxf(sqrtf(ss), 1e-12f);

  float aa = sigmoidf_(aio[i0] + a_bias[d0]);
  float ab = sigmoidf_(aio[i1] + a_bias[d1]);

  float kk0 = kha * inv, kk1 = khb * inv;
  aio[i0] = -kk0;            aio[i1] = -kk1;
  bterm[i0] = kk0 * aa;      bterm[i1] = kk1 * ab;
  kio[i0] = k0a * (1.0f + (aa - 1.0f) * kaw[d0]);
  kio[i1] = k0b * (1.0f + (ab - 1.0f) * kaw[d1]);
  wio[i0] = __expf(LOG_HALF * sigmoidf_(wio[i0] + w_bias[d0]));
  wio[i1] = __expf(LOG_HALF * sigmoidf_(wio[i1] + w_bias[d1]));
}

// ---------------------------------------------------------------------------
// Sequential delta-rule scan. One block per (b,h); 64 threads; thread owns
// state column v (64 regs). Next timestep's operands prefetched into regs
// while the two unrolled FMA passes run.
// ---------------------------------------------------------------------------
__global__ __launch_bounds__(64)
void scan_kernel(const float* __restrict__ r,  const float* __restrict__ ew,
                 const float* __restrict__ kf, const float* __restrict__ at,
                 const float* __restrict__ bt, const float* __restrict__ v,
                 float* __restrict__ o) {
  __shared__ __align__(16) float l_r[64], l_ew[64], l_kf[64], l_at[64], l_bt[64];
  int bh = blockIdx.x;
  int b = bh / H_, h = bh - b * H_;
  int tid = threadIdx.x;
  size_t base = (size_t)b * T_ * D_ + (size_t)h * 64;

  float S[64];
#pragma unroll
  for (int k = 0; k < 64; ++k) S[k] = 0.0f;

  float p_r = r[base + tid],  p_ew = ew[base + tid], p_kf = kf[base + tid];
  float p_at = at[base + tid], p_bt = bt[base + tid], p_v = v[base + tid];

  for (int t = 0; t < T_; ++t) {
    l_r [tid] = p_r;  l_ew[tid] = p_ew; l_kf[tid] = p_kf;
    l_at[tid] = p_at; l_bt[tid] = p_bt;
    float myv = p_v;
    __syncthreads();

    if (t + 1 < T_) {                    // prefetch next step
      size_t nidx = base + (size_t)(t + 1) * D_ + tid;
      p_r = r[nidx];  p_ew = ew[nidx]; p_kf = kf[nidx];
      p_at = at[nidx]; p_bt = bt[nidx]; p_v = v[nidx];
    }

    float s0 = 0.f, s1 = 0.f, s2 = 0.f, s3 = 0.f;
#pragma unroll
    for (int k = 0; k < 64; k += 4) {
      float4 b4 = *reinterpret_cast<const float4*>(l_bt + k);
      s0 += b4.x * S[k];     s1 += b4.y * S[k + 1];
      s2 += b4.z * S[k + 2]; s3 += b4.w * S[k + 3];
    }
    float sab = (s0 + s1) + (s2 + s3);

    float o0 = 0.f, o1 = 0.f, o2 = 0.f, o3 = 0.f;
#pragma unroll
    for (int k = 0; k < 64; k += 4) {
      float4 e4 = *reinterpret_cast<const float4*>(l_ew + k);
      float4 a4 = *reinterpret_cast<const float4*>(l_at + k);
      float4 k4 = *reinterpret_cast<const float4*>(l_kf + k);
      float4 r4 = *reinterpret_cast<const float4*>(l_r  + k);
      S[k]     = e4.x * S[k]     + a4.x * sab + k4.x * myv;  o0 += r4.x * S[k];
      S[k + 1] = e4.y * S[k + 1] + a4.y * sab + k4.y * myv;  o1 += r4.y * S[k + 1];
      S[k + 2] = e4.z * S[k + 2] + a4.z * sab + k4.z * myv;  o2 += r4.z * S[k + 2];
      S[k + 3] = e4.w * S[k + 3] + a4.w * sab + k4.w * myv;  o3 += r4.w * S[k + 3];
    }
    o[base + (size_t)t * D_ + tid] = (o0 + o1) + (o2 + o3);
    __syncthreads();
  }
}

// ---------------------------------------------------------------------------
// GroupNorm(num_groups=H) + bonus + gate -> bf16 GEMM operand.
// ---------------------------------------------------------------------------
__global__ __launch_bounds__(256)
void gn_kernel(const float* __restrict__ o,  const float* __restrict__ r,
               const float* __restrict__ kf, const float* __restrict__ v,
               const float* __restrict__ g,  const float* __restrict__ r_k,
               const float* __restrict__ gnw, const float* __restrict__ gnb,
               __hip_bfloat16* __restrict__ yb) {
  int lane = threadIdx.x & 31;
  int wid  = threadIdx.x >> 5;
  long row = (long)blockIdx.x * 8 + wid;     // over B*T*H
  int  h   = (int)(row % H_);
  size_t e0 = (size_t)row * 64;
  size_t i0 = e0 + lane, i1 = i0 + 32;
  int d0 = h * 64 + lane, d1 = d0 + 32;

  float oa = o[i0], ob = o[i1];
  float sum = oa + ob;
#pragma unroll
  for (int m = 16; m >= 1; m >>= 1) sum += __shfl_xor(sum, m, 32);
  float mu = sum * (1.0f / 64.0f);
  float ca = oa - mu, cb = ob - mu;
  float vs = ca * ca + cb * cb;
#pragma unroll
  for (int m = 16; m >= 1; m >>= 1) vs += __shfl_xor(vs, m, 32);
  float inv = rsqrtf(vs * (1.0f / 64.0f) + EPS_GN);

  float dot = r[i0] * kf[i0] * r_k[d0] + r[i1] * kf[i1] * r_k[d1];
#pragma unroll
  for (int m = 16; m >= 1; m >>= 1) dot += __shfl_xor(dot, m, 32);

  float ona = ca * inv * gnw[d0] + gnb[d0];
  float onb = cb * inv * gnw[d1] + gnb[d1];
  float ya  = (ona + dot * v[i0]) * g[i0];
  float yb_ = (onb + dot * v[i1]) * g[i1];
  yb[i0] = __float2bfloat16(ya);
  yb[i1] = __float2bfloat16(yb_);
}

// ---------------------------------------------------------------------------
extern "C" void kernel_launch(void* const* d_in, const int* in_sizes, int n_in,
                              void* d_out, int out_size, void* d_ws, size_t ws_size,
                              hipStream_t stream) {
  const float* q    = (const float*)d_in[0];
  const float* hs   = (const float*)d_in[1];
  /* d_in[2] = x_r mix: reference computes xr but r uses raw query -> unused */
  const float* mw   = (const float*)d_in[3];
  const float* mk   = (const float*)d_in[4];
  const float* mv   = (const float*)d_in[5];
  const float* ma   = (const float*)d_in[6];
  const float* mg   = (const float*)d_in[7];
  const float* k_k  = (const float*)d_in[8];
  const float* k_a  = (const float*)d_in[9];
  const float* r_k  = (const float*)d_in[10];
  const float* W_r  = (const float*)d_in[11];
  const float* W_k  = (const float*)d_in[12];
  const float* W_v  = (const float*)d_in[13];
  const float* W_o  = (const float*)d_in[14];
  const float* wla  = (const float*)d_in[15];
  const float* wlb  = (const float*)d_in[16];
  const float* wlbs = (const float*)d_in[17];
  const float* ala  = (const float*)d_in[18];
  const float* alb  = (const float*)d_in[19];
  const float* albs = (const float*)d_in[20];
  const float* gla  = (const float*)d_in[21];
  const float* glb  = (const float*)d_in[22];
  const float* gnw  = (const float*)d_in[23];
  const float* gnb  = (const float*)d_in[24];
  float* out = (float*)d_out;
  (void)in_sizes; (void)n_in; (void)out_size; (void)ws_size;

  constexpr size_t SZ = (size_t)M_ * D_;           // 8388608 elements
  char* ws = (char*)d_ws;
  size_t off = 0;
  auto alloc = [&](size_t bytes) -> char* {
    char* p = ws + off;
    off += (bytes + 255) & ~(size_t)255;
    return p;
  };

  __hip_bfloat16* qb  = (__hip_bfloat16*)alloc(SZ * 2);
  __hip_bfloat16* xwb = (__hip_bfloat16*)alloc(SZ * 2);
  __hip_bfloat16* xkb = (__hip_bfloat16*)alloc(SZ * 2);
  __hip_bfloat16* xvb = (__hip_bfloat16*)alloc(SZ * 2);
  __hip_bfloat16* xab = (__hip_bfloat16*)alloc(SZ * 2);
  __hip_bfloat16* xgb = (__hip_bfloat16*)alloc(SZ * 2);
  __hip_bfloat16* WrT  = (__hip_bfloat16*)alloc((size_t)D_ * D_ * 2);
  __hip_bfloat16* WkT  = (__hip_bfloat16*)alloc((size_t)D_ * D_ * 2);
  __hip_bfloat16* WvT  = (__hip_bfloat16*)alloc((size_t)D_ * D_ * 2);
  __hip_bfloat16* WoT  = (__hip_bfloat16*)alloc((size_t)D_ * D_ * 2);
  __hip_bfloat16* wlaT = (__hip_bfloat16*)alloc((size_t)64 * D_ * 2);
  __hip_bfloat16* wlbT = (__hip_bfloat16*)alloc((size_t)D_ * 64 * 2);
  __hip_bfloat16* alaT = (__hip_bfloat16*)alloc((size_t)64 * D_ * 2);
  __hip_bfloat16* albT = (__hip_bfloat16*)alloc((size_t)D_ * 64 * 2);
  __hip_bfloat16* glaT = (__hip_bfloat16*)alloc((size_t)128 * D_ * 2);
  __hip_bfloat16* glbT = (__hip_bfloat16*)alloc((size_t)D_ * 128 * 2);
  float* r_f  = (float*)alloc(SZ * 4);
  float* k_f  = (float*)alloc(SZ * 4);   // k0 -> kfin (in place)
  float* v_f  = (float*)alloc(SZ * 4);
  float* w1   = (float*)alloc((size_t)M_ * 64 * 4);
  float* a1   = (float*)alloc((size_t)M_ * 64 * 4);
  float* g1   = (float*)alloc((size_t)M_ * 128 * 4);
  __hip_bfloat16* t1w = (__hip_bfloat16*)alloc((size_t)M_ * 64 * 2);
  __hip_bfloat16* t1a = (__hip_bfloat16*)alloc((size_t)M_ * 64 * 2);
  __hip_bfloat16* t1g = (__hip_bfloat16*)alloc((size_t)M_ * 128 * 2);
  float* w2   = (float*)alloc(SZ * 4);   // -> ew (in place)
  float* a2   = (float*)alloc(SZ * 4);   // -> aterm (in place)
  float* gval = (float*)alloc(SZ * 4);
  float* btrm = (float*)alloc(SZ * 4);
  float* o_f  = (float*)alloc(SZ * 4);
  __hip_bfloat16* y_b = (__hip_bfloat16*)alloc(SZ * 2);

  auto cdiv = [](size_t a, size_t b) { return (unsigned)((a + b - 1) / b); };

  // 1) weight convert/transpose
  transcvt_kernel<<<cdiv((size_t)D_*D_,256),256,0,stream>>>(W_r, WrT, D_, D_);
  transcvt_kernel<<<cdiv((size_t)D_*D_,256),256,0,stream>>>(W_k, WkT, D_, D_);
  transcvt_kernel<<<cdiv((size_t)D_*D_,256),256,0,stream>>>(W_v, WvT, D_, D_);
  transcvt_kernel<<<cdiv((size_t)D_*D_,256),256,0,stream>>>(W_o, WoT, D_, D_);
  transcvt_kernel<<<cdiv((size_t)D_*64 ,256),256,0,stream>>>(wla, wlaT, D_, 64);
  transcvt_kernel<<<cdiv((size_t)64*D_ ,256),256,0,stream>>>(wlb, wlbT, 64, D_);
  transcvt_kernel<<<cdiv((size_t)D_*64 ,256),256,0,stream>>>(ala, alaT, D_, 64);
  transcvt_kernel<<<cdiv((size_t)64*D_ ,256),256,0,stream>>>(alb, albT, 64, D_);
  transcvt_kernel<<<cdiv((size_t)D_*128,256),256,0,stream>>>(gla, glaT, D_, 128);
  transcvt_kernel<<<cdiv((size_t)128*D_,256),256,0,stream>>>(glb, glbT, 128, D_);

  // 2) mixing
  mix_kernel<<<cdiv(SZ,256),256,0,stream>>>(hs, q, mw, mk, mv, ma, mg,
                                            qb, xwb, xkb, xvb, xab, xgb);

  // 3) stage-1 GEMMs
  dim3 blk(256);
  gemm_bf16_nt<<<dim3(M_/128, D_/128), blk, 0, stream>>>(qb,  WrT,  r_f, M_, D_,  D_);
  gemm_bf16_nt<<<dim3(M_/128, D_/128), blk, 0, stream>>>(xkb, WkT,  k_f, M_, D_,  D_);
  gemm_bf16_nt<<<dim3(M_/128, D_/128), blk, 0, stream>>>(xvb, WvT,  v_f, M_, D_,  D_);
  gemm_bf16_nt<<<dim3(M_/128, 1),      blk, 0, stream>>>(xwb, wlaT, w1,  M_, 64,  D_);
  gemm_bf16_nt<<<dim3(M_/128, 1),      blk, 0, stream>>>(xab, alaT, a1,  M_, 64,  D_);
  gemm_bf16_nt<<<dim3(M_/128, 1),      blk, 0, stream>>>(xgb, glaT, g1,  M_, 128, D_);

  // 4) LoRA mid activations
  act_cvt_kernel<<<cdiv((size_t)M_*64 ,256),256,0,stream>>>(w1, t1w, (size_t)M_*64, 1);
  act_cvt_kernel<<<cdiv((size_t)M_*64 ,256),256,0,stream>>>(a1, t1a, (size_t)M_*64, 0);
  act_cvt_kernel<<<cdiv((size_t)M_*128,256),256,0,stream>>>(g1, t1g, (size_t)M_*128, 2);

  // 5) stage-2 LoRA GEMMs
  gemm_bf16_nt<<<dim3(M_/128, D_/128), blk, 0, stream>>>(t1w, wlbT, w2,   M_, D_, 64);
  gemm_bf16_nt<<<dim3(M_/128, D_/128), blk, 0, stream>>>(t1a, albT, a2,   M_, D_, 64);
  gemm_bf16_nt<<<dim3(M_/128, D_/128), blk, 0, stream>>>(t1g, glbT, gval, M_, D_, 128);

  // 6) prep (k->kfin, a2->aterm, w2->ew, bterm)
  prep_kernel<<<(B_*T_*H_)/8, 256, 0, stream>>>(k_f, a2, w2, btrm, k_k, k_a, albs, wlbs);

  // 7) sequential scan
  scan_kernel<<<B_*H_, 64, 0, stream>>>(r_f, w2, k_f, a2, btrm, v_f, o_f);

  // 8) GroupNorm + bonus + gate
  gn_kernel<<<(B_*T_*H_)/8, 256, 0, stream>>>(o_f, r_f, k_f, v_f, gval, r_k, gnw, gnb, y_b);

  // 9) output projection
  gemm_bf16_nt<<<dim3(M_/128, D_/128), blk, 0, stream>>>(y_b, WoT, out, M_, D_, D_);
}
